// SpatialGCN_56444460204637
// MI455X (gfx1250) — compile-verified, matrix operand order checked by reading, not run on hardware
//
#include <hip/hip_runtime.h>
#include <hip/hip_bf16.h>

typedef __attribute__((ext_vector_type(16))) __bf16 v16bf;
typedef __attribute__((ext_vector_type(8)))  float  v8f;

__device__ __forceinline__ __bf16 f2bf(float f) {
  union { float f; unsigned u; } v; v.f = f;
  unsigned r = (v.u + 0x7FFFu + ((v.u >> 16) & 1u)) >> 16;  // RNE
  unsigned short s = (unsigned short)r;
  __bf16 out; __builtin_memcpy(&out, &s, sizeof(out));
  return out;
}

// ---------------- elementwise / graph-prep kernels ----------------

__global__ void gcn_zero_f32(float* __restrict__ p, long long n) {
  long long i = (long long)blockIdx.x * blockDim.x + threadIdx.x;
  if (i < n) p[i] = 0.f;
}

__global__ void gcn_deg_scatter(const float* __restrict__ w, const int* __restrict__ col,
                                float* __restrict__ deg, int E) {
  int e = blockIdx.x * blockDim.x + threadIdx.x;
  if (e < E) atomicAdd(&deg[col[e]], w[e]);
}

__global__ void gcn_rsqrt_inplace(float* __restrict__ deg, int n) {
  int i = blockIdx.x * blockDim.x + threadIdx.x;
  if (i < n) { float d = deg[i]; deg[i] = (d > 0.f) ? rsqrtf(d) : 0.f; }
}

__global__ void gcn_edge_norm(const float* __restrict__ dis, const float* __restrict__ w,
                              const int* __restrict__ row, const int* __restrict__ col,
                              float* __restrict__ nrm, int E) {
  int e = blockIdx.x * blockDim.x + threadIdx.x;
  if (e < E) nrm[e] = dis[row[e]] * w[e] * dis[col[e]];
}

// f32 -> bf16 row-major streaming convert (for A matrices)
__global__ void gcn_f32_to_bf16(const float* __restrict__ src, __bf16* __restrict__ dst,
                                long long n) {
  long long i = (long long)blockIdx.x * blockDim.x + threadIdx.x;
  if (i < n) dst[i] = f2bf(src[i]);
}

// Pack B (row-major KxN f32) into fragment-ready bf16: [ntile][kstep][lane][16]
// lane = half*16 + n16 ; element i<8 -> k = ks*32 + half*8 + i ; i>=8 -> +16
__global__ void gcn_pack_b(const float* __restrict__ B, __bf16* __restrict__ Bsw,
                           int K, int Nc) {
  int idx = blockIdx.x * blockDim.x + threadIdx.x;
  if (idx >= K * Nc) return;
  int i    = idx & 15;
  int lane = (idx >> 4) & 31;
  int t    = idx >> 9;               // ntile*KSTEPS + ks
  int KS   = K >> 5;
  int ks   = t % KS;
  int nt   = t / KS;
  int half = lane >> 4;
  int n    = nt * 16 + (lane & 15);
  int k    = ks * 32 + half * 8 + ((i < 8) ? i : (16 + (i - 8)));
  Bsw[idx] = f2bf(B[(size_t)k * Nc + n]);
}

// ---------------- WMMA bf16 GEMM: D(MxN) = A(MxK) * Bsw ----------------
// Block: 256 threads = 8 waves; block tile 32(M) x 128(N); wave owns two
// 16x16 tiles (rows 0-15 / 16-31) sharing one B fragment -> two independent
// WMMA accumulator chains interleave to hide WMMA->WMMA hazard slots, and
// B global traffic is halved. A staged in LDS (b128), B pre-swizzled (b128).

template <int K, int Nc>
__global__ __launch_bounds__(256)
void gcn_gemm_bf16(const __bf16* __restrict__ A, const __bf16* __restrict__ Bsw,
                   float* __restrict__ D, int M) {
  constexpr int KP     = K + 8;            // stride in bf16, multiple of 8 (16B)
  constexpr int KSTEPS = K / 32;
  __shared__ __attribute__((aligned(16))) __bf16 As[32 * KP];

  const int mBase = blockIdx.x * 32;

  // Cooperative A tile load (32 x K bf16), 16B chunks per thread.
  constexpr int CH = K / 64;               // chunks of 8 bf16 per thread
#pragma unroll
  for (int c = 0; c < CH; ++c) {
    int l = (c * 256 + (int)threadIdx.x) * 8;
    int m = l / K, k = l % K;
    int g = mBase + m;
    uint4 v = make_uint4(0u, 0u, 0u, 0u);
    if (g < M) v = *(const uint4*)(A + (size_t)g * K + k);
    *(uint4*)(&As[m * KP + k]) = v;
  }
  __syncthreads();

  const int wave  = threadIdx.x >> 5;
  const int lane  = threadIdx.x & 31;
  const int half  = lane >> 4;
  const int l16   = lane & 15;
  const int ntile = blockIdx.y * 8 + wave;          // 16-wide N tile index
  const int nCol  = ntile * 16 + l16;

  const __bf16* bbase = Bsw + ((size_t)ntile * KSTEPS) * 512 + lane * 16;

  v8f acc0 = {}, acc1 = {};
#pragma unroll
  for (int ks = 0; ks < KSTEPS; ++ks) {
    const int k0 = ks * 32 + half * 8;
    union { v16bf v; uint4 q[2]; } a0, a1, bu;
    a0.q[0] = *(const uint4*)(&As[l16 * KP + k0]);
    a0.q[1] = *(const uint4*)(&As[l16 * KP + k0 + 16]);
    a1.q[0] = *(const uint4*)(&As[(l16 + 16) * KP + k0]);
    a1.q[1] = *(const uint4*)(&As[(l16 + 16) * KP + k0 + 16]);
    bu.q[0] = *(const uint4*)(bbase + (size_t)ks * 512);
    bu.q[1] = *(const uint4*)(bbase + (size_t)ks * 512 + 8);
    acc0 = __builtin_amdgcn_wmma_f32_16x16x32_bf16(
        false, a0.v, false, bu.v, (short)0, acc0, false, false);
    acc1 = __builtin_amdgcn_wmma_f32_16x16x32_bf16(
        false, a1.v, false, bu.v, (short)0, acc1, false, false);
  }

  // D 16x16 f32: VGPR r -> M = r + 8*half, N = lane%16 (Nc constexpr ->
  // row offsets fold into immediate offsets).
  float* d0 = D + (size_t)(mBase + 8 * half) * Nc + nCol;
  float* d1 = D + (size_t)(mBase + 16 + 8 * half) * Nc + nCol;
  if (mBase + 32 <= M) {
#pragma unroll
    for (int r = 0; r < 8; ++r) d0[(size_t)r * Nc] = acc0[r];
#pragma unroll
    for (int r = 0; r < 8; ++r) d1[(size_t)r * Nc] = acc1[r];
  } else {
#pragma unroll
    for (int r = 0; r < 8; ++r)
      if (mBase + 8 * half + r < M) d0[(size_t)r * Nc] = acc0[r];
#pragma unroll
    for (int r = 0; r < 8; ++r)
      if (mBase + 16 + 8 * half + r < M) d1[(size_t)r * Nc] = acc1[r];
  }
}

// ---------------- edge gather-scale-scatter (atomic) ----------------
// One thread handles 4 features of one edge: out[col] += h[row] * norm.

template <int F>
__global__ void gcn_scatter_edges(const float* __restrict__ H, const float* __restrict__ nrm,
                                  const int* __restrict__ row, const int* __restrict__ col,
                                  float* __restrict__ out, int E) {
  constexpr int PER = F >> 2;              // float4 groups per edge (power of 2)
  int idx = blockIdx.x * blockDim.x + threadIdx.x;
  int e = idx / PER;                       // constexpr pow2 -> shift
  if (e >= E) return;
  int f = (idx & (PER - 1)) << 2;
  int r = row[e], c = col[e];
  float s = nrm[e];
  const float4 h4 = *(const float4*)(H + (size_t)r * F + f);
  float* o = out + (size_t)c * F + f;
  atomicAdd(o + 0, h4.x * s);
  atomicAdd(o + 1, h4.y * s);
  atomicAdd(o + 2, h4.z * s);
  atomicAdd(o + 3, h4.w * s);
}

// relu(agg + b) -> bf16 (feeds layer-2 WMMA GEMM directly)
__global__ void gcn_bias_relu_bf16(const float* __restrict__ agg, const float* __restrict__ b,
                                   __bf16* __restrict__ out, long long n, int F) {
  long long i = (long long)blockIdx.x * blockDim.x + threadIdx.x;
  if (i < n) {
    float v = agg[i] + b[(int)(i % F)];
    out[i] = f2bf(v > 0.f ? v : 0.f);
  }
}

__global__ void gcn_bias_add(float* __restrict__ p, const float* __restrict__ b,
                             long long n, int F) {
  long long i = (long long)blockIdx.x * blockDim.x + threadIdx.x;
  if (i < n) p[i] += b[(int)(i % F)];
}

// ---------------- launch ----------------

extern "C" void kernel_launch(void* const* d_in, const int* in_sizes, int n_in,
                              void* d_out, int out_size, void* d_ws, size_t ws_size,
                              hipStream_t stream) {
  const float* x  = (const float*)d_in[0];
  const int*   ei = (const int*)d_in[1];
  const float* ew = (const float*)d_in[2];
  const float* W1 = (const float*)d_in[3];
  const float* b1 = (const float*)d_in[4];
  const float* W2 = (const float*)d_in[5];
  const float* b2 = (const float*)d_in[6];

  constexpr int IN = 128, HID = 256, OUT = 128;
  const int Nn = in_sizes[0] / IN;     // 100000 nodes
  const int E  = in_sizes[2];          // 800000 edges
  const int* row = ei;
  const int* col = ei + E;

  // workspace carve-out (256B-aligned regions)
  char* wp = (char*)d_ws;
  auto carve = [&](size_t bytes) -> void* {
    void* r = (void*)wp;
    wp += (bytes + 255) & ~(size_t)255;
    return r;
  };
  float*  deg  = (float*) carve((size_t)Nn * 4);
  float*  nrm  = (float*) carve((size_t)E * 4);
  float*  h1   = (float*) carve((size_t)Nn * HID * 4);  // layer-1 GEMM out (f32)
  float*  agg  = (float*) carve((size_t)Nn * HID * 4);  // layer-1 aggregate (f32)
  __bf16* xbf  = (__bf16*)carve((size_t)Nn * IN * 2);   // bf16 copy of x
  __bf16* W1sw = (__bf16*)carve((size_t)IN * HID * 2);  // swizzled bf16 W1
  __bf16* W2sw = (__bf16*)carve((size_t)HID * OUT * 2); // swizzled bf16 W2
  // h1 region reuse after scatter1: first half = hbf (bf16), second half = h2 (f32)
  __bf16* hbf  = (__bf16*)h1;
  float*  h2   = h1 + (size_t)Nn * (HID / 2);
  float*  out  = (float*)d_out;

  const dim3 blk(256);
  const long long nh = (long long)Nn * HID;
  const long long no = (long long)Nn * OUT;
  auto blks = [](long long n) { return (unsigned)((n + 255) / 256); };

  // normalization: deg -> dis -> per-edge norm
  gcn_zero_f32      <<<blks(Nn), blk, 0, stream>>>(deg, Nn);
  gcn_deg_scatter   <<<blks(E),  blk, 0, stream>>>(ew, col, deg, E);
  gcn_rsqrt_inplace <<<blks(Nn), blk, 0, stream>>>(deg, Nn);
  gcn_edge_norm     <<<blks(E),  blk, 0, stream>>>(deg, ew, row, col, nrm, E);

  // one-time precision/layout prep
  gcn_f32_to_bf16 <<<blks((long long)Nn * IN), blk, 0, stream>>>(x, xbf, (long long)Nn * IN);
  gcn_pack_b      <<<blks(IN * HID),  blk, 0, stream>>>(W1, W1sw, IN, HID);
  gcn_pack_b      <<<blks(HID * OUT), blk, 0, stream>>>(W2, W2sw, HID, OUT);

  // layer 1: h1 = x @ W1 ; agg = scatter(norm * h1[row] -> col) ; relu(agg+b1) -> bf16
  dim3 g1((unsigned)((Nn + 31) / 32), (unsigned)(HID / 128));
  gcn_gemm_bf16<IN, HID> <<<g1, blk, 0, stream>>>(xbf, W1sw, h1, Nn);
  gcn_zero_f32 <<<blks(nh), blk, 0, stream>>>(agg, nh);
  gcn_scatter_edges<HID> <<<blks((long long)E * (HID / 4)), blk, 0, stream>>>(
      h1, nrm, row, col, agg, E);
  gcn_bias_relu_bf16 <<<blks(nh), blk, 0, stream>>>(agg, b1, hbf, nh, HID);

  // layer 2: h2 = relu_bf16 @ W2 ; out = scatter(norm * h2[row] -> col) + b2
  dim3 g2((unsigned)((Nn + 31) / 32), (unsigned)(OUT / 128));
  gcn_gemm_bf16<HID, OUT> <<<g2, blk, 0, stream>>>(hbf, W2sw, h2, Nn);
  gcn_zero_f32 <<<blks(no), blk, 0, stream>>>(out, no);
  gcn_scatter_edges<OUT> <<<blks((long long)E * (OUT / 4)), blk, 0, stream>>>(
      h2, nrm, row, col, out, E);
  gcn_bias_add <<<blks(no), blk, 0, stream>>>(out, b2, no, OUT);
}
